// HierarchicalSparseAttention_31911607009442
// MI455X (gfx1250) — matrix operand order repaired and checked by reading, not compiled, not run on hardware
//
#include <hip/hip_runtime.h>

// ---------------------------------------------------------------------------
// HierarchicalSparseAttention for MI455X (gfx1250, wave32, WMMA)
// B=2, N=2048, D=1024, H=16, Dh=64. GEMMs in bf16 WMMA (f32 accum),
// hierarchical 12-way attention in f32 VALU with wave32 shuffles.
// Round 3: 64x64 per-wave tile (4 A frags x 4 B frags -> 16 wmma / K-step,
// 1.0 b128-load per wmma = 16 B/wmma of cache traffic, vs 24 B/wmma before).
// ---------------------------------------------------------------------------

#define DMODEL 1024
#define NSEQ   2048
#define XYROWS 4096   // 2*N-1 = 4095 rows, padded to 4096 (pad row zeroed)
#define NBATCH 2
#define NHEADS 16
#define NLOG   12     // self + 11 hierarchical neighbors (log2(2048) = 11)

typedef __attribute__((ext_vector_type(16))) __bf16 v16bf;
typedef __attribute__((ext_vector_type(8)))  float  v8f;

__device__ __forceinline__ unsigned short f2bf(float f) {
  unsigned u = __float_as_uint(f);
  u += 0x7FFFu + ((u >> 16) & 1u);           // round-to-nearest-even
  return (unsigned short)(u >> 16);
}
__device__ __forceinline__ float bf2f(unsigned hbits) {
  return __uint_as_float(hbits << 16);
}

union FragBF {
  v16bf v;
  uint4 q[2];
};

// ---------------------------------------------------------------------------
// Fuse concat([query, y], axis=1) with fp32 -> bf16 conversion; pad row 4095.
// ---------------------------------------------------------------------------
__global__ __launch_bounds__(256) void pack_xy_bf16(
    const float* __restrict__ q, const float* __restrict__ y,
    unsigned short* __restrict__ xy)
{
  const long long i = (long long)blockIdx.x * blockDim.x + threadIdx.x;
  const int c = (int)(i % DMODEL);
  const long long rowid = i / DMODEL;
  const int r = (int)(rowid % XYROWS);
  const int b = (int)(rowid / XYROWS);
  float v;
  if (r < NSEQ)            v = q[((long long)b * NSEQ + r) * DMODEL + c];
  else if (r < XYROWS - 1) v = y[((long long)b * (NSEQ - 1) + (r - NSEQ)) * DMODEL + c];
  else                     v = 0.0f;
  xy[i] = f2bf(v);
}

__global__ __launch_bounds__(256) void convert_f32_bf16(
    const float* __restrict__ src, unsigned short* __restrict__ dst, int n)
{
  const int i = blockIdx.x * blockDim.x + threadIdx.x;
  if (i < n) dst[i] = f2bf(src[i]);
}

// ---------------------------------------------------------------------------
// C[M,1024] = A_bf16[M,1024] @ W_bf16[1024,1024]^T   (both K-contiguous)
// Block: 256 threads = 8 waves arranged 2(M) x 4(N); block tile 128x256.
// Each wave owns a 64(M) x 64(N) strip: 4 A fragments x 4 B fragments ->
// 16 C tiles, 16 wmma per K-step of 32, one b128 load per wmma.
// Fragment layouts per CDNA5 ISA 7.12.2 (16-bit A 16x32, B 32x16, f32 C).
// Output: bf16 (Cbf) when Cf==nullptr, else f32 with bias.
// ---------------------------------------------------------------------------
__global__ __launch_bounds__(256) void gemm_bf16_wmma(
    const unsigned short* __restrict__ A, long long strideA,
    const unsigned short* __restrict__ W,
    unsigned short* __restrict__ Cbf,
    float* __restrict__ Cf, const float* __restrict__ bias,
    long long strideC)
{
  const int wave   = threadIdx.x >> 5;
  const int lane   = threadIdx.x & 31;
  const int lane15 = lane & 15;
  const bool hi    = lane >= 16;

  const long long b  = blockIdx.z;
  const int m0 = blockIdx.y * 128 + (wave >> 2) * 64;
  const int n0 = blockIdx.x * 256 + (wave & 3) * 64;

  // A fragment rows for this lane: M = m0 + 16*mi + lane15,
  // K chunks {k0+base .. +7} and {k0+base+16 .. +23}, base = hi ? 8 : 0.
  // mi sub-tiles live at immediate offsets 0/32768/65536/98304 bytes.
  const unsigned short* arow =
      A + b * strideA + (long long)(m0 + lane15) * DMODEL + (hi ? 8 : 0);
  // B fragment base: column N = n0 + 16t + lane15, K chunks {k0+base .. +7} and {k0+base+16 .. +23}, base = hi ? 8 : 0
  const unsigned short* wbase =
      W + (long long)(n0 + lane15) * DMODEL + (hi ? 8 : 0);

  const v8f vzero = {0.f, 0.f, 0.f, 0.f, 0.f, 0.f, 0.f, 0.f};
  v8f acc[4][4] = {{vzero, vzero, vzero, vzero}, {vzero, vzero, vzero, vzero},
                   {vzero, vzero, vzero, vzero}, {vzero, vzero, vzero, vzero}};

  #pragma unroll 1
  for (int k0 = 0; k0 < DMODEL; k0 += 32) {
    // Pull the streamed A operand 4 K-steps ahead (speculative prefetch ->
    // global_prefetch_b8); lanes cover all 64 rows via the 4 sub-tiles.
    // W is block/L2-resident by reuse.
    __builtin_prefetch(arow + 0 * 16 * DMODEL + k0 + 128, 0, 1);
    __builtin_prefetch(arow + 1 * 16 * DMODEL + k0 + 128, 0, 1);
    __builtin_prefetch(arow + 2 * 16 * DMODEL + k0 + 128, 0, 1);
    __builtin_prefetch(arow + 3 * 16 * DMODEL + k0 + 128, 0, 1);

    FragBF a[4];
    #pragma unroll
    for (int mi = 0; mi < 4; ++mi) {
      const unsigned short* ap = arow + (long long)(mi * 16) * DMODEL + k0;
      a[mi].q[0] = *(const uint4*)(ap);
      a[mi].q[1] = *(const uint4*)(ap + 16);
    }

    #pragma unroll
    for (int t = 0; t < 4; ++t) {
      const unsigned short* wrow = wbase + (long long)(t * 16) * DMODEL + k0;
      FragBF bf;
      bf.q[0] = *(const uint4*)(wrow);
      bf.q[1] = *(const uint4*)(wrow + 16);
      #pragma unroll
      for (int mi = 0; mi < 4; ++mi) {
        acc[mi][t] = __builtin_amdgcn_wmma_f32_16x16x32_bf16(
            false, a[mi].v, false, bf.v, (short)0, acc[mi][t], false, false);
      }
    }
  }

  // C 16x16 f32 layout: VGPR r -> row base + (hi?8:0) + r, col = n0+16t+lane15
  #pragma unroll
  for (int mi = 0; mi < 4; ++mi) {
    const int rowbase = m0 + mi * 16 + (hi ? 8 : 0);
    #pragma unroll
    for (int t = 0; t < 4; ++t) {
      const int col = n0 + t * 16 + lane15;
      #pragma unroll
      for (int r = 0; r < 8; ++r) {
        const long long off = b * strideC + (long long)(rowbase + r) * DMODEL + col;
        const float v = acc[mi][t][r];
        if (Cf) Cf[off] = v + bias[col];
        else    Cbf[off] = f2bf(v);
      }
    }
  }
}

// ---------------------------------------------------------------------------
// Hierarchical sparse attention: one wave per (b, h, n).
// 12 logits = self + 11 tree neighbors; indices recomputed in-register:
//   idx0 = n ^ 1; idx_l = ((idx_{l-1} >> 1) + N) ^ 1  (matches reference).
// Lane covers Dh elements {2*lane, 2*lane+1}; butterfly __shfl_xor reductions.
// ---------------------------------------------------------------------------
__global__ __launch_bounds__(256) void hier_attn(
    const unsigned short* __restrict__ Qb,   // [B, 2048, 1024] bf16
    const unsigned short* __restrict__ Kb,   // [B, 4096, 1024] bf16
    const unsigned short* __restrict__ Vb,   // [B, 4096, 1024] bf16
    unsigned short* __restrict__ Ob)         // [B, 2048, 1024] bf16
{
  const int lane = threadIdx.x & 31;
  const int gw   = blockIdx.x * 8 + (threadIdx.x >> 5);
  const int n = gw & (NSEQ - 1);
  const int h = (gw >> 11) & (NHEADS - 1);
  const int b = gw >> 15;

  int rows[NLOG];
  rows[0] = n;
  unsigned cur = (unsigned)n ^ 1u;
  rows[1] = (int)cur;
  #pragma unroll
  for (int l = 2; l < NLOG; ++l) {
    cur = ((cur >> 1) + (unsigned)NSEQ) ^ 1u;
    rows[l] = (int)cur;
  }

  const int col = h * 64 + 2 * lane;
  const long long qoff = ((long long)b * NSEQ + n) * DMODEL + col;
  const unsigned qp = *(const unsigned*)(Qb + qoff);
  const float q0 = bf2f(qp & 0xffffu), q1 = bf2f(qp >> 16);

  const long long kvbase = (long long)b * XYROWS * DMODEL;

  float logit[NLOG];
  #pragma unroll
  for (int j = 0; j < NLOG; ++j) {
    const unsigned kp = *(const unsigned*)(Kb + kvbase + (long long)rows[j] * DMODEL + col);
    float p = q0 * bf2f(kp & 0xffffu) + q1 * bf2f(kp >> 16);
    #pragma unroll
    for (int o = 16; o > 0; o >>= 1) p += __shfl_xor(p, o, 32);
    logit[j] = p * 0.125f;                    // scale = 1/sqrt(64)
  }

  float m = logit[0];
  #pragma unroll
  for (int j = 1; j < NLOG; ++j) m = fmaxf(m, logit[j]);
  float w[NLOG], s = 0.f;
  #pragma unroll
  for (int j = 0; j < NLOG; ++j) { w[j] = __expf(logit[j] - m); s += w[j]; }
  const float inv = 1.0f / s;

  float a0 = 0.f, a1 = 0.f;
  #pragma unroll
  for (int j = 0; j < NLOG; ++j) {
    const unsigned vp = *(const unsigned*)(Vb + kvbase + (long long)rows[j] * DMODEL + col);
    const float wj = w[j] * inv;
    a0 += wj * bf2f(vp & 0xffffu);
    a1 += wj * bf2f(vp >> 16);
  }

  const unsigned out = (unsigned)f2bf(a0) | ((unsigned)f2bf(a1) << 16);
  *(unsigned*)(Ob + qoff) = out;
}

// ---------------------------------------------------------------------------
// Launch: inputs = {query,key,value,y,Wq,Wk,Wv,Wo,bo}; key/value unused
// (reference projects K/V from xy, not from the raw key/value tensors).
// ---------------------------------------------------------------------------
extern "C" void kernel_launch(void* const* d_in, const int* in_sizes, int n_in,
                              void* d_out, int out_size, void* d_ws, size_t ws_size,
                              hipStream_t stream) {
  (void)in_sizes; (void)n_in; (void)out_size; (void)ws_size;

  const float* query = (const float*)d_in[0];
  const float* y     = (const float*)d_in[3];
  const float* Wq    = (const float*)d_in[4];
  const float* Wk    = (const float*)d_in[5];
  const float* Wv    = (const float*)d_in[6];
  const float* Wo    = (const float*)d_in[7];
  const float* bo    = (const float*)d_in[8];

  char* ws = (char*)d_ws;
  size_t off = 0;
  auto carve = [&](size_t bytes) -> char* {
    off = (off + 255) & ~(size_t)255;
    char* p = ws + off;
    off += bytes;
    return p;
  };

  unsigned short* xyb = (unsigned short*)carve((size_t)NBATCH * XYROWS * DMODEL * 2);
  unsigned short* wqb = (unsigned short*)carve((size_t)DMODEL * DMODEL * 2);
  unsigned short* wkb = (unsigned short*)carve((size_t)DMODEL * DMODEL * 2);
  unsigned short* wvb = (unsigned short*)carve((size_t)DMODEL * DMODEL * 2);
  unsigned short* wob = (unsigned short*)carve((size_t)DMODEL * DMODEL * 2);
  unsigned short* qb  = (unsigned short*)carve((size_t)NBATCH * NSEQ   * DMODEL * 2);
  unsigned short* kb  = (unsigned short*)carve((size_t)NBATCH * XYROWS * DMODEL * 2);
  unsigned short* vb  = (unsigned short*)carve((size_t)NBATCH * XYROWS * DMODEL * 2);
  unsigned short* ab  = (unsigned short*)carve((size_t)NBATCH * NSEQ   * DMODEL * 2);

  // 1) pack & convert inputs to bf16
  {
    const long long total = (long long)NBATCH * XYROWS * DMODEL;
    pack_xy_bf16<<<(unsigned)(total / 256), 256, 0, stream>>>(query, y, xyb);
  }
  const int wsz = DMODEL * DMODEL;
  convert_f32_bf16<<<wsz / 256, 256, 0, stream>>>(Wq, wqb, wsz);
  convert_f32_bf16<<<wsz / 256, 256, 0, stream>>>(Wk, wkb, wsz);
  convert_f32_bf16<<<wsz / 256, 256, 0, stream>>>(Wv, wvb, wsz);
  convert_f32_bf16<<<wsz / 256, 256, 0, stream>>>(Wo, wob, wsz);

  const long long strideXY = (long long)XYROWS * DMODEL;
  const long long strideN  = (long long)NSEQ * DMODEL;

  // 2) Q = query @ Wq.T  (query == first 2048 rows of xy)
  {
    dim3 g(DMODEL / 256, NSEQ / 128, NBATCH);
    gemm_bf16_wmma<<<g, 256, 0, stream>>>(xyb, strideXY, wqb, qb, nullptr, nullptr, strideN);
  }
  // 3) Kf = xy @ Wk.T ; Vf = xy @ Wv.T   (4096 padded rows)
  {
    dim3 g(DMODEL / 256, XYROWS / 128, NBATCH);
    gemm_bf16_wmma<<<g, 256, 0, stream>>>(xyb, strideXY, wkb, kb, nullptr, nullptr, strideXY);
    gemm_bf16_wmma<<<g, 256, 0, stream>>>(xyb, strideXY, wvb, vb, nullptr, nullptr, strideXY);
  }
  // 4) hierarchical attention (one wave per (b,h,n))
  {
    const int waves = NBATCH * NHEADS * NSEQ;   // 65536
    hier_attn<<<waves / 8, 256, 0, stream>>>(qb, kb, vb, ab);
  }
  // 5) out = attn @ Wo.T + bo  (f32 output)
  {
    dim3 g(DMODEL / 256, NSEQ / 128, NBATCH);
    gemm_bf16_wmma<<<g, 256, 0, stream>>>(ab, strideN, wob, nullptr, (float*)d_out, bo, strideN);
  }
}
